// StaticExpansionBlock_3058016715061
// MI455X (gfx1250) — matrix operand order repaired and checked by reading, not compiled, not run on hardware
//
#include <hip/hip_runtime.h>
#include <cstddef>

// ---------------------------------------------------------------------------
// StaticExpansionBlock for MI455X (gfx1250, wave32, WMMA).
// All contractions use v_wmma_f32_16x16x32_bf16 (bf16 operands, f32 accum).
// Shared A-tiles are staged in LDS via GLOBAL_LOAD_ASYNC_TO_LDS_B128 when the
// toolchain exposes the gfx1250 builtins (plain copies otherwise).
//
// Shapes: B=16, E=2048, D=512, T=992 (=31*32), NSEG=5,
//         segment starts {0,32,96,224,480} (all multiples of 32).
//
// WMMA operand conventions (CDNA5 ISA 7.12.2):
//  A (16x32 bf16, MxK): lane l holds row m=l&15; elems 0..7  = A[m][k0+(l>>4)*8 ..]
//                                                elems 8..15 = A[m][k0+16+(l>>4)*8 ..]
//  B (32x16 bf16, KxN): lane l holds col n=l&15; elems 0..7  = B[k0+(l>>4)*8 ..][n]
//                                                elems 8..15 = B[k0+16+(l>>4)*8 ..][n]
//      -> loaded from Bt stored N x K row-major (two runs of 8 contiguous bf16).
//  C/D (16x16 f32): VGPR r, lanes 0-15: (m=r, n=lane); lanes 16-31: (m=8+r, n=lane-16)
//      -> 8 consecutive M rows per lane/column => contiguous bf16x8 transposed store.
// ---------------------------------------------------------------------------

#define B_   16
#define E_   2048
#define DME  512
#define T_   992
#define NSEG 5

#if defined(__AMDGCN__) && __has_builtin(__builtin_amdgcn_global_load_async_to_lds_b128) && __has_builtin(__builtin_amdgcn_s_wait_asynccnt)
#define USE_ASYNC_LDS 1
#else
#define USE_ASYNC_LDS 0
#endif

typedef __attribute__((ext_vector_type(16))) __bf16 bf16x16;
typedef __attribute__((ext_vector_type(8)))  __bf16 bf16x8;
typedef __attribute__((ext_vector_type(8)))  float  f32x8;
typedef int v4i_t __attribute__((vector_size(16)));

__device__ inline unsigned short f2bfu(float f){
  union { float f; unsigned u; } v; v.f = f;
  unsigned r = v.u + 0x7FFFu + ((v.u >> 16) & 1u);   // round-to-nearest-even
  return (unsigned short)(r >> 16);
}
__device__ inline __bf16 f2bf(float f){
  union { unsigned short s; __bf16 b; } o; o.s = f2bfu(f); return o.b;
}
__device__ inline float bf2f(__bf16 b){
  union { unsigned short s; __bf16 b; } i; i.b = b;
  union { unsigned u; float f; } o; o.u = ((unsigned)i.s) << 16; return o.f;
}
__device__ inline bf16x16 pack16(bf16x8 lo, bf16x8 hi){
  bf16x16 r;
#pragma unroll
  for (int i = 0; i < 8; ++i){ r[i] = lo[i]; r[i+8] = hi[i]; }
  return r;
}
// A fragment from row-major MxK bf16 (works for global or LDS pointers).
__device__ inline bf16x16 load_a16(const __bf16* A, int lda, int m, int k0, int half){
  const __bf16* p = A + (size_t)m * lda + (size_t)(k0 + half * 8);
  bf16x8 lo = *(const bf16x8*)p;
  bf16x8 hi = *(const bf16x8*)(p + 16);
  return pack16(lo, hi);
}
// B fragment from N x K row-major bf16 (n is this lane's column).
__device__ inline bf16x16 load_b16(const __bf16* Bt, int ldb, int n, int k0, int half){
  const __bf16* p = Bt + (size_t)n * ldb + (size_t)(k0 + half * 8);
  bf16x8 lo = *(const bf16x8*)p;
  bf16x8 hi = *(const bf16x8*)(p + 16);
  return pack16(lo, hi);
}
#define WMMA_BF16(a, b, c) \
  __builtin_amdgcn_wmma_f32_16x16x32_bf16(false, (a), false, (b), (short)0, (c), false, false)

__device__ inline int seg_of(int t){
  return (t < 32) ? 0 : (t < 96) ? 1 : (t < 224) ? 2 : (t < 480) ? 3 : 4;
}
__device__ inline float pick5(float v0, float v1, float v2, float v3, float v4, int s){
  return (s == 0) ? v0 : (s == 1) ? v1 : (s == 2) ? v2 : (s == 3) ? v3 : v4;
}

// 16-byte global -> LDS stage (async path when available).
// Builtin signature: (v4i addrspace(1)*, v4i addrspace(3)*, imm offset, imm cpol).
__device__ inline void stage_b128(const __bf16* g, __bf16* l){
#if USE_ASYNC_LDS
  __builtin_amdgcn_global_load_async_to_lds_b128(
      (__attribute__((address_space(1))) v4i_t*)g,
      (__attribute__((address_space(3))) v4i_t*)l,
      0, 0);
#else
  *(uint4*)l = *(const uint4*)g;
#endif
}
__device__ inline void stage_fence(){
#if USE_ASYNC_LDS
  __builtin_amdgcn_s_wait_asynccnt(0);
#endif
  __syncthreads();
}

// ---------------------------------------------------------------------------
// Elementwise kernels
// ---------------------------------------------------------------------------
__global__ void __launch_bounds__(256) k_cvt_bf16(const float* __restrict__ src,
                                                  unsigned short* __restrict__ dst, int n){
  int i = blockIdx.x * 256 + threadIdx.x;
  if (i < n) dst[i] = f2bfu(src[i]);
}

// q_bf16[b,t,d] = bf16(query_table[n_indexes[b,t], d])
__global__ void __launch_bounds__(256) k_gather_q(const float* __restrict__ qt,
                                                  const int* __restrict__ nidx,
                                                  unsigned short* __restrict__ q, int n){
  int i = blockIdx.x * 256 + threadIdx.x;
  if (i < n){
    int bt = i >> 9;              // D = 512
    int d  = i & (DME - 1);
    q[i] = f2bfu(qt[(size_t)nidx[bt] * DME + d]);
  }
}

// ---------------------------------------------------------------------------
// Projection GEMM: C = x @ W^T + bias.  M = B*E = 32768, N = K = 512.
// Wave tile 64x64 (4x4). mode: 0 = bf16 [M,N] (Wk),
// 1 = bf16 transposed per batch [B][N][E] (Wa/Wb), 2 = sigmoid f32 [M,N] (Ws).
// Grid: 4096 waves = 512 blocks x 8 waves (exact).
// ---------------------------------------------------------------------------
__global__ void __launch_bounds__(256) k_proj_gemm(const unsigned short* __restrict__ A_,
                                                   const unsigned short* __restrict__ W_,
                                                   const float* __restrict__ bias,
                                                   void* __restrict__ out, int mode){
  const __bf16* A = (const __bf16*)A_;
  const __bf16* W = (const __bf16*)W_;
  const int lane = threadIdx.x & 31;
  const int wid  = blockIdx.x * 8 + (threadIdx.x >> 5);
  const int m0 = (wid >> 3) * 64, n0 = (wid & 7) * 64;
  const int row = lane & 15, half = lane >> 4;

  f32x8 acc[4][4];
#pragma unroll
  for (int i = 0; i < 4; ++i)
#pragma unroll
    for (int j = 0; j < 4; ++j)
#pragma unroll
      for (int q = 0; q < 8; ++q) acc[i][j][q] = 0.f;

  for (int k0 = 0; k0 < DME; k0 += 32){
    if (k0 + 32 < DME){
      __builtin_prefetch(A + (size_t)(m0 + row) * DME + k0 + 32, 0, 0);
      __builtin_prefetch(W + (size_t)(n0 + row) * DME + k0 + 32, 0, 0);
    }
    bf16x16 af[4], bfm[4];
#pragma unroll
    for (int i = 0; i < 4; ++i) af[i]  = load_a16(A, DME, m0 + i * 16 + row, k0, half);
#pragma unroll
    for (int j = 0; j < 4; ++j) bfm[j] = load_b16(W, DME, n0 + j * 16 + row, k0, half);
#pragma unroll
    for (int i = 0; i < 4; ++i)
#pragma unroll
      for (int j = 0; j < 4; ++j)
        acc[i][j] = WMMA_BF16(af[i], bfm[j], acc[i][j]);
  }

#pragma unroll
  for (int i = 0; i < 4; ++i){
    const int mB = m0 + i * 16 + half * 8;     // first of 8 consecutive M rows
#pragma unroll
    for (int j = 0; j < 4; ++j){
      const int n = n0 + j * 16 + row;
      const float bv = bias[n];
      if (mode == 0){
        __bf16* o = (__bf16*)out;
#pragma unroll
        for (int q = 0; q < 8; ++q) o[(size_t)(mB + q) * DME + n] = f2bf(acc[i][j][q] + bv);
      } else if (mode == 1){
        const int bb = mB >> 11;               // E = 2048
        const int e0 = mB & (E_ - 1);
        bf16x8 v;
#pragma unroll
        for (int q = 0; q < 8; ++q) v[q] = f2bf(acc[i][j][q] + bv);
        *(bf16x8*)((__bf16*)out + ((size_t)bb * DME + n) * E_ + e0) = v;
      } else {
        float* o = (float*)out;
#pragma unroll
        for (int q = 0; q < 8; ++q){
          float z = acc[i][j][q] + bv;
          o[(size_t)(mB + q) * DME + n] = 1.0f / (1.0f + __expf(-z));
        }
      }
    }
  }
}

// ---------------------------------------------------------------------------
// z GEMM: z[b,t,e] = (q[b,t,:] . xk[b,e,:]) / sqrt(D).
// Per batch: M=T=992 (31 tiles of 32), N=E=2048 (16 tiles of 128), K=512.
// All 8 waves of a block share the same 32 Q rows -> staged once in LDS (32 KB).
// Grid: 16 * 31 * 16 = 7936 waves = 992 blocks (exact; 62 blocks per batch).
// Stores z as bf16 [B][T][E] and zT as bf16 [B][E][T].
// ---------------------------------------------------------------------------
__global__ void __launch_bounds__(256) k_z_gemm(const unsigned short* __restrict__ Q_,
                                                const unsigned short* __restrict__ XK_,
                                                unsigned short* __restrict__ Z_,
                                                unsigned short* __restrict__ ZT_){
  __shared__ __bf16 sQ[32 * DME];                       // 32 KB
  const int lane = threadIdx.x & 31;
  const int wv   = threadIdx.x >> 5;
  const int b    = blockIdx.x / 62;                     // 62 blocks per batch
  const int rblk = (blockIdx.x % 62) * 8;
  const int tm   = rblk >> 4;                           // uniform across block
  const int tn   = (rblk & 15) + wv;
  const int m0   = tm * 32, n0 = tn * 128;
  const int row  = lane & 15, half = lane >> 4;

  const __bf16* Qg = (const __bf16*)Q_ + ((size_t)b * T_ + m0) * DME;
  const __bf16* XK = (const __bf16*)XK_ + (size_t)b * E_ * DME;

  // cooperative stage of the shared 32x512 Q tile (contiguous in memory)
  for (int pc = threadIdx.x; pc < 2048; pc += 256)
    stage_b128(Qg + pc * 8, sQ + pc * 8);
  stage_fence();

  f32x8 acc[2][8];
#pragma unroll
  for (int i = 0; i < 2; ++i)
#pragma unroll
    for (int j = 0; j < 8; ++j)
#pragma unroll
      for (int q = 0; q < 8; ++q) acc[i][j][q] = 0.f;

  for (int k0 = 0; k0 < DME; k0 += 32){
    if (k0 + 32 < DME)
      __builtin_prefetch(XK + (size_t)(n0 + row) * DME + k0 + 32, 0, 0);
    bf16x16 af[2];
#pragma unroll
    for (int i = 0; i < 2; ++i) af[i] = load_a16(sQ, DME, i * 16 + row, k0, half);
#pragma unroll
    for (int j = 0; j < 8; ++j){
      bf16x16 bfm = load_b16(XK, DME, n0 + j * 16 + row, k0, half);
#pragma unroll
      for (int i = 0; i < 2; ++i) acc[i][j] = WMMA_BF16(af[i], bfm, acc[i][j]);
    }
  }

  __bf16* Z  = (__bf16*)Z_  + (size_t)b * T_ * E_;
  __bf16* ZT = (__bf16*)ZT_ + (size_t)b * E_ * T_;
  const float sc = 0.04419417382415922f;       // 1/sqrt(512)
#pragma unroll
  for (int i = 0; i < 2; ++i){
    const int t0 = m0 + i * 16 + half * 8;
#pragma unroll
    for (int j = 0; j < 8; ++j){
      const int e = n0 + j * 16 + row;
      bf16x8 v;
#pragma unroll
      for (int q = 0; q < 8; ++q){
        float zv = acc[i][j][q] * sc;
        __bf16 zb = f2bf(zv);
        Z[(size_t)(t0 + q) * E_ + e] = zb;
        v[q] = zb;
      }
      *(bf16x8*)(ZT + (size_t)e * T_ + t0) = v;  // transposed tile store
    }
  }
}

// ---------------------------------------------------------------------------
// Forward denominators: per (b,t), suma = sum_e relu(z)*mask, sumb = sum_e relu(-z)*mask.
// ---------------------------------------------------------------------------
__global__ void __launch_bounds__(256) k_fw_den(const unsigned short* __restrict__ Z_,
                                                const int* __restrict__ mask,
                                                float* __restrict__ invA,
                                                float* __restrict__ invB){
  const int bt = blockIdx.x;                   // b*T + t
  const int b  = bt / T_;
  const __bf16* zr = (const __bf16*)Z_ + (size_t)bt * E_;
  float sa = 0.f, sb = 0.f;
  for (int e = threadIdx.x; e < E_; e += 256){
    if (mask[b * E_ + e]){
      float v = bf2f(zr[e]);
      sa += fmaxf(v, 0.f);
      sb += fmaxf(-v, 0.f);
    }
  }
  __shared__ float sA[256], sB[256];
  sA[threadIdx.x] = sa; sB[threadIdx.x] = sb;
  __syncthreads();
  for (int s = 128; s > 0; s >>= 1){
    if ((int)threadIdx.x < s){ sA[threadIdx.x] += sA[threadIdx.x + s]; sB[threadIdx.x] += sB[threadIdx.x + s]; }
    __syncthreads();
  }
  if (threadIdx.x == 0){
    invA[bt] = 1.f / (sA[0] + 1e-9f);
    invB[bt] = 1.f / (sB[0] + 1e-9f);
  }
}

// ---------------------------------------------------------------------------
// Backward per-segment denominators: per (b,e,seg), sum over t-in-seg of relu(+/-zT).
// ---------------------------------------------------------------------------
__global__ void __launch_bounds__(256) k_bw_den(const unsigned short* __restrict__ ZT_,
                                                float* __restrict__ inv5A,
                                                float* __restrict__ inv5B){
  const __bf16* zr = (const __bf16*)ZT_ + (size_t)blockIdx.x * T_;   // b*E + e
  float la0=0,la1=0,la2=0,la3=0,la4=0, lb0=0,lb1=0,lb2=0,lb3=0,lb4=0;
  for (int t = threadIdx.x; t < T_; t += 256){
    float v = bf2f(zr[t]);
    float ra = fmaxf(v, 0.f), rb = fmaxf(-v, 0.f);
    int s = seg_of(t);
    if      (s == 0){ la0 += ra; lb0 += rb; }
    else if (s == 1){ la1 += ra; lb1 += rb; }
    else if (s == 2){ la2 += ra; lb2 += rb; }
    else if (s == 3){ la3 += ra; lb3 += rb; }
    else            { la4 += ra; lb4 += rb; }
  }
  __shared__ float gA[NSEG], gB[NSEG];
  if (threadIdx.x < NSEG){ gA[threadIdx.x] = 0.f; gB[threadIdx.x] = 0.f; }
  __syncthreads();
  atomicAdd(&gA[0], la0); atomicAdd(&gA[1], la1); atomicAdd(&gA[2], la2);
  atomicAdd(&gA[3], la3); atomicAdd(&gA[4], la4);
  atomicAdd(&gB[0], lb0); atomicAdd(&gB[1], lb1); atomicAdd(&gB[2], lb2);
  atomicAdd(&gB[3], lb3); atomicAdd(&gB[4], lb4);
  __syncthreads();
  if (threadIdx.x < NSEG){
    size_t o = (size_t)blockIdx.x * NSEG + threadIdx.x;
    inv5A[o] = 1.f / (gA[threadIdx.x] + 1e-9f);
    inv5B[o] = 1.f / (gB[threadIdx.x] + 1e-9f);
  }
}

// ---------------------------------------------------------------------------
// Class GEMM: class_a[t,d] = sum_e (relu(z[t,e])*invA[t]) * xa[e,d] + bias_exp.
// Per batch: M=T=992 (31 tiles of 32), N=D=512 (8 tiles of 64), K=E=2048.
// All 8 waves share the same 32 z rows -> staged in LDS in 512-wide K chunks (32 KB).
// Grid: 16 * 31 * 8 = 3968 waves = 496 blocks (exact; 31 per batch).
// Outputs stored transposed: caT/cbT [B][D][T].
// ---------------------------------------------------------------------------
__global__ void __launch_bounds__(256) k_class_gemm(const unsigned short* __restrict__ Z_,
                                                    const unsigned short* __restrict__ XAT_,
                                                    const unsigned short* __restrict__ XBT_,
                                                    const float* __restrict__ invA,
                                                    const float* __restrict__ invB,
                                                    const int* __restrict__ nidx,
                                                    const float* __restrict__ bias_table,
                                                    unsigned short* __restrict__ CAT_,
                                                    unsigned short* __restrict__ CBT_){
  __shared__ __bf16 sZ[32 * 512];                       // 32 KB K-chunk of z
  const int lane = threadIdx.x & 31;
  const int wv   = threadIdx.x >> 5;
  const int b    = blockIdx.x / 31;                     // 31 blocks per batch
  const int tm   = blockIdx.x % 31;                     // uniform across block
  const int m0   = tm * 32, n0 = wv * 64;
  const int row  = lane & 15, half = lane >> 4;

  const __bf16* Zg = (const __bf16*)Z_   + ((size_t)b * T_ + m0) * E_;
  const __bf16* XA = (const __bf16*)XAT_ + (size_t)b * DME * E_;
  const __bf16* XB = (const __bf16*)XBT_ + (size_t)b * DME * E_;

  float ia[2], ib[2];
#pragma unroll
  for (int i = 0; i < 2; ++i){
    int t = m0 + i * 16 + row;
    ia[i] = invA[b * T_ + t];
    ib[i] = invB[b * T_ + t];
  }

  f32x8 accA[2][4], accB[2][4];
#pragma unroll
  for (int i = 0; i < 2; ++i)
#pragma unroll
    for (int j = 0; j < 4; ++j)
#pragma unroll
      for (int q = 0; q < 8; ++q){ accA[i][j][q] = 0.f; accB[i][j][q] = 0.f; }

  for (int kc = 0; kc < E_; kc += 512){
    // stage z rows m0..m0+31, cols kc..kc+511 (2048 x 16B pieces)
    __syncthreads();
    for (int pc = threadIdx.x; pc < 2048; pc += 256){
      int rr = pc >> 6;            // piece / 64
      int c8 = (pc & 63) * 8;
      stage_b128(Zg + (size_t)rr * E_ + kc + c8, sZ + rr * 512 + c8);
    }
    stage_fence();

    for (int k0 = kc; k0 < kc + 512; k0 += 32){
      __builtin_prefetch(XA + (size_t)(n0 + row) * E_ + k0 + 32, 0, 0);
      __builtin_prefetch(XB + (size_t)(n0 + row) * E_ + k0 + 32, 0, 0);
      bf16x16 fa[2], fb[2];
#pragma unroll
      for (int i = 0; i < 2; ++i){
        bf16x16 zf = load_a16(sZ, 512, i * 16 + row, k0 - kc, half);
#pragma unroll
        for (int q = 0; q < 16; ++q){
          float v = bf2f(zf[q]);
          fa[i][q] = f2bf(fmaxf(v, 0.f)  * ia[i]);
          fb[i][q] = f2bf(fmaxf(-v, 0.f) * ib[i]);
        }
      }
#pragma unroll
      for (int j = 0; j < 4; ++j){
        bf16x16 ba = load_b16(XA, E_, n0 + j * 16 + row, k0, half);
        bf16x16 bb = load_b16(XB, E_, n0 + j * 16 + row, k0, half);
#pragma unroll
        for (int i = 0; i < 2; ++i){
          accA[i][j] = WMMA_BF16(fa[i], ba, accA[i][j]);
          accB[i][j] = WMMA_BF16(fb[i], bb, accB[i][j]);
        }
      }
    }
  }

  __bf16* CAT = (__bf16*)CAT_ + (size_t)b * DME * T_;
  __bf16* CBT = (__bf16*)CBT_ + (size_t)b * DME * T_;
#pragma unroll
  for (int i = 0; i < 2; ++i){
    const int t0 = m0 + i * 16 + half * 8;
#pragma unroll
    for (int j = 0; j < 4; ++j){
      const int d = n0 + j * 16 + row;
      bf16x8 va, vb;
#pragma unroll
      for (int q = 0; q < 8; ++q){
        int t = t0 + q;
        float bias = bias_table[(size_t)nidx[b * T_ + t] * DME + d];
        va[q] = f2bf(accA[i][j][q] + bias);
        vb[q] = f2bf(accB[i][j][q] + bias);
      }
      *(bf16x8*)(CAT + (size_t)d * T_ + t0) = va;
      *(bf16x8*)(CBT + (size_t)d * T_ + t0) = vb;
    }
  }
}

// ---------------------------------------------------------------------------
// Backward GEMM + final combine:
// out[e,d] = sel*(sum_t a_bw[e,t]*ca[t,d])/5 + (1-sel)*(sum_t b_bw[e,t]*cb[t,d])/5
// Per batch: M=E=2048 (64 tiles of 32), N=D=512 (8 tiles of 64), K=T=992.
// All 8 waves share the same 32 zT rows -> full K staged in LDS (62 KB).
// Segment boundaries are multiples of 32 -> segment scale uniform per K step.
// Grid: 16 * 64 * 8 = 8192 waves = 1024 blocks (exact; 64 per batch).
// ---------------------------------------------------------------------------
__global__ void __launch_bounds__(256) k_back_gemm(const unsigned short* __restrict__ ZT_,
                                                   const unsigned short* __restrict__ CAT_,
                                                   const unsigned short* __restrict__ CBT_,
                                                   const float* __restrict__ inv5A,
                                                   const float* __restrict__ inv5B,
                                                   const float* __restrict__ sel,
                                                   float* __restrict__ out){
  __shared__ __bf16 sZT[32 * T_];                       // 62 KB (full K)
  const int lane = threadIdx.x & 31;
  const int wv   = threadIdx.x >> 5;
  const int b    = blockIdx.x / 64;                     // 64 blocks per batch
  const int tm   = blockIdx.x % 64;                     // uniform across block
  const int m0   = tm * 32, n0 = wv * 64;
  const int row  = lane & 15, half = lane >> 4;

  const __bf16* ZTg = (const __bf16*)ZT_  + ((size_t)b * E_ + m0) * T_;
  const __bf16* CA  = (const __bf16*)CAT_ + (size_t)b * DME * T_;
  const __bf16* CB  = (const __bf16*)CBT_ + (size_t)b * DME * T_;

  // stage zT rows m0..m0+31 (32 x 992 bf16 = 3968 x 16B pieces)
  for (int pc = threadIdx.x; pc < 32 * 124; pc += 256){
    int rr = pc / 124;
    int c8 = (pc % 124) * 8;
    stage_b128(ZTg + (size_t)rr * T_ + c8, sZT + rr * T_ + c8);
  }
  stage_fence();

  float ia[2][NSEG], ib[2][NSEG];
#pragma unroll
  for (int i = 0; i < 2; ++i){
    size_t base = ((size_t)b * E_ + (m0 + i * 16 + row)) * NSEG;
#pragma unroll
    for (int s = 0; s < NSEG; ++s){ ia[i][s] = inv5A[base + s]; ib[i][s] = inv5B[base + s]; }
  }

  f32x8 accA[2][4], accB[2][4];
#pragma unroll
  for (int i = 0; i < 2; ++i)
#pragma unroll
    for (int j = 0; j < 4; ++j)
#pragma unroll
      for (int q = 0; q < 8; ++q){ accA[i][j][q] = 0.f; accB[i][j][q] = 0.f; }

  for (int k0 = 0; k0 < T_; k0 += 32){
    const int sg = seg_of(k0);
    bf16x16 fa[2], fb[2];
#pragma unroll
    for (int i = 0; i < 2; ++i){
      float sca = pick5(ia[i][0], ia[i][1], ia[i][2], ia[i][3], ia[i][4], sg);
      float scb = pick5(ib[i][0], ib[i][1], ib[i][2], ib[i][3], ib[i][4], sg);
      bf16x16 zf = load_a16(sZT, T_, i * 16 + row, k0, half);
#pragma unroll
      for (int q = 0; q < 16; ++q){
        float v = bf2f(zf[q]);
        fa[i][q] = f2bf(fmaxf(v, 0.f)  * sca);
        fb[i][q] = f2bf(fmaxf(-v, 0.f) * scb);
      }
    }
#pragma unroll
    for (int j = 0; j < 4; ++j){
      bf16x16 ca = load_b16(CA, T_, n0 + j * 16 + row, k0, half);
      bf16x16 cb = load_b16(CB, T_, n0 + j * 16 + row, k0, half);
#pragma unroll
      for (int i = 0; i < 2; ++i){
        accA[i][j] = WMMA_BF16(fa[i], ca, accA[i][j]);
        accB[i][j] = WMMA_BF16(fb[i], cb, accB[i][j]);
      }
    }
  }

  const float invn = 0.2f;                     // 1 / n_seg
#pragma unroll
  for (int i = 0; i < 2; ++i){
    const int e0 = m0 + i * 16 + half * 8;
#pragma unroll
    for (int j = 0; j < 4; ++j){
      const int d = n0 + j * 16 + row;
#pragma unroll
      for (int q = 0; q < 8; ++q){
        size_t off = ((size_t)b * E_ + (e0 + q)) * DME + d;
        float s = sel[off];
        out[off] = (s * accA[i][j][q] + (1.f - s) * accB[i][j][q]) * invn;
      }
    }
  }
}

// ---------------------------------------------------------------------------
// Host launcher
// ---------------------------------------------------------------------------
extern "C" void kernel_launch(void* const* d_in, const int* in_sizes, int n_in,
                              void* d_out, int out_size, void* d_ws, size_t ws_size,
                              hipStream_t stream){
  const float* x   = (const float*)d_in[0];
  const int*   idx = (const int*)  d_in[1];
  const int*   msk = (const int*)  d_in[2];
  const float* qt  = (const float*)d_in[3];
  const float* bt  = (const float*)d_in[4];
  const float* Wk  = (const float*)d_in[5];
  const float* bk  = (const float*)d_in[6];
  const float* Wa  = (const float*)d_in[7];
  const float* ba  = (const float*)d_in[8];
  const float* Wb  = (const float*)d_in[9];
  const float* bb  = (const float*)d_in[10];
  const float* Ws  = (const float*)d_in[11];
  const float* bs  = (const float*)d_in[12];
  float* out = (float*)d_out;

  // Workspace carve-up (~384 MB total).
  char* p = (char*)d_ws;
  size_t off = 0;
  auto take = [&](size_t bytes) -> char* {
    char* r = p + off;
    off = (off + bytes + 255) & ~(size_t)255;
    return r;
  };
  const size_t BED = (size_t)B_ * E_ * DME;
  const size_t BTD = (size_t)B_ * T_ * DME;
  const size_t BTE = (size_t)B_ * T_ * E_;
  const size_t DD  = (size_t)DME * DME;

  unsigned short* x_bf  = (unsigned short*)take(BED * 2);
  unsigned short* wk_bf = (unsigned short*)take(DD * 2);
  unsigned short* wa_bf = (unsigned short*)take(DD * 2);
  unsigned short* wb_bf = (unsigned short*)take(DD * 2);
  unsigned short* ws_bf = (unsigned short*)take(DD * 2);
  unsigned short* q_bf  = (unsigned short*)take(BTD * 2);
  unsigned short* xk_bf = (unsigned short*)take(BED * 2);
  unsigned short* xaT   = (unsigned short*)take(BED * 2);
  unsigned short* xbT   = (unsigned short*)take(BED * 2);
  float*          selb  = (float*)         take(BED * 4);
  unsigned short* z_bf  = (unsigned short*)take(BTE * 2);
  unsigned short* zT_bf = (unsigned short*)take(BTE * 2);
  unsigned short* caT   = (unsigned short*)take(BTD * 2);
  unsigned short* cbT   = (unsigned short*)take(BTD * 2);
  float*          invAf = (float*)take((size_t)B_ * T_ * 4);
  float*          invBf = (float*)take((size_t)B_ * T_ * 4);
  float*          inv5A = (float*)take((size_t)B_ * E_ * NSEG * 4);
  float*          inv5B = (float*)take((size_t)B_ * E_ * NSEG * 4);
  (void)ws_size; (void)in_sizes; (void)n_in; (void)out_size;

  // 1) bf16 conversions + gathers
  k_cvt_bf16<<<(int)((BED + 255) / 256), 256, 0, stream>>>(x,  x_bf,  (int)BED);
  k_cvt_bf16<<<(int)((DD  + 255) / 256), 256, 0, stream>>>(Wk, wk_bf, (int)DD);
  k_cvt_bf16<<<(int)((DD  + 255) / 256), 256, 0, stream>>>(Wa, wa_bf, (int)DD);
  k_cvt_bf16<<<(int)((DD  + 255) / 256), 256, 0, stream>>>(Wb, wb_bf, (int)DD);
  k_cvt_bf16<<<(int)((DD  + 255) / 256), 256, 0, stream>>>(Ws, ws_bf, (int)DD);
  k_gather_q<<<(int)((BTD + 255) / 256), 256, 0, stream>>>(qt, idx, q_bf, (int)BTD);

  // 2) projections: 4096 waves -> 512 blocks of 8 waves (exact)
  k_proj_gemm<<<512, 256, 0, stream>>>(x_bf, wk_bf, bk, (void*)xk_bf, 0);
  k_proj_gemm<<<512, 256, 0, stream>>>(x_bf, wa_bf, ba, (void*)xaT,  1);
  k_proj_gemm<<<512, 256, 0, stream>>>(x_bf, wb_bf, bb, (void*)xbT,  1);
  k_proj_gemm<<<512, 256, 0, stream>>>(x_bf, ws_bf, bs, (void*)selb, 2);

  // 3) z GEMM: 992 blocks (exact)
  k_z_gemm<<<992, 256, 0, stream>>>(q_bf, xk_bf, z_bf, zT_bf);

  // 4) denominators
  k_fw_den<<<B_ * T_, 256, 0, stream>>>(z_bf, msk, invAf, invBf);
  k_bw_den<<<B_ * E_, 256, 0, stream>>>(zT_bf, inv5A, inv5B);

  // 5) class GEMM: 496 blocks (exact)
  k_class_gemm<<<496, 256, 0, stream>>>(z_bf, xaT, xbT, invAf, invBf, idx, bt, caT, cbT);

  // 6) backward GEMM + combine: 1024 blocks (exact)
  k_back_gemm<<<1024, 256, 0, stream>>>(zT_bf, caT, cbT, inv5A, inv5B, selb, out);
}
